// RNNCFModel_36653250904162
// MI455X (gfx1250) — matrix operand order repaired and checked by reading, not compiled, not run on hardware
//
#include <hip/hip_runtime.h>
#include <hip/hip_bf16.h>

typedef __attribute__((ext_vector_type(16))) _Float16 v16h;
typedef __attribute__((ext_vector_type(8)))  float    v8f;

union V16H { unsigned u[8]; v16h h; };

__device__ __forceinline__ unsigned pk(float a, float b) {
    union { _Float16 h[2]; unsigned u; } t;
    t.h[0] = (_Float16)a; t.h[1] = (_Float16)b;
    return t.u;
}
__device__ __forceinline__ float sigm(float x) {
    return __builtin_amdgcn_rcpf(1.0f + __expf(-x));
}
__device__ __forceinline__ float tanh_f(float x) {
    return 2.0f * __builtin_amdgcn_rcpf(1.0f + __expf(-2.0f * x)) - 1.0f;
}
__device__ __forceinline__ float nanfix(float x) { return (x != x) ? 1.0f : x; }

// One wave32 per block; wave owns 16 vehicles and runs the full 1000-step recurrence.
// GEMMs are computed transposed: z^T(80x16) = Wcat^T(80x32) @ X(32x16) with
// X rows 0..11 = preprocessed inputs, rows 12..31 = h.  A operands (weights)
// stay resident in VGPRs; C is seeded with the bias.
__global__ __launch_bounds__(32) void rnncf_kernel(
    const float* __restrict__ inp,        // [nveh][nt][12]
    const float* __restrict__ init_state, // [nveh][2]
    const float* __restrict__ h0,         // [nveh][20]
    const float* __restrict__ c0,         // [nveh][20]
    const float* __restrict__ lk,         // [12][80]
    const float* __restrict__ lrk,        // [20][80]
    const float* __restrict__ lb,         // [80]
    const float* __restrict__ d2w,        // [20][10]
    const float* __restrict__ d2b,        // [10]
    const float* __restrict__ lcw,        // [10][3]
    const float* __restrict__ lcb,        // [3]
    const float* __restrict__ d1w,        // [10][1]
    const float* __restrict__ d1b,        // [1]
    float* __restrict__ out_pos,          // [nveh][nt]
    float* __restrict__ out_lc,           // [nveh][nt][3]
    float* __restrict__ out_spd,          // [nveh]
    float* __restrict__ out_h,            // [nveh][20]
    float* __restrict__ out_c,            // [nveh][20]
    int nveh, int nt)
{
    // Per-wave LDS slab (single-wave workgroup -> DS ops are in-order, no barrier)
    __shared__ unsigned Blds[32][8];  // packed f16 B operand, 32x16, contiguous k = 16*half + element
    __shared__ float    zlds[16][80]; // z^T staging: [vehicle][80]
    __shared__ float    xlds[16][16]; // relu(x) staging: [vehicle][unit(padded)]

    const int lane = threadIdx.x & 31;
    const int hi   = lane >> 4;       // lane half
    const int v    = lane & 15;       // local vehicle / column
    const int vg   = blockIdx.x * 16 + v;
    const int ub   = 10 * hi;         // this lane owns LSTM units ub..ub+9 of vehicle v

    // ---- Build constant A tiles (f16, A 16x32, contiguous k order) ------------
    // A lane-half shifts K by 16; word w holds K pair k0,k0+1 with
    // k0 = 16*hi + 2*w.  M = lane&15.
    V16H Az[5], A2, A3;
    v8f  biasZ[5], bias2, bias3;
#pragma unroll
    for (int w = 0; w < 8; ++w) {
        const int k0 = 16 * hi + 2 * w;
#pragma unroll
        for (int m = 0; m < 5; ++m) {
            const int j = 16 * m + v;
            const float lo = (k0     < 12) ? lk[k0 * 80 + j]       : lrk[(k0 - 12) * 80 + j];
            const float hif= (k0 + 1 < 12) ? lk[(k0 + 1) * 80 + j] : lrk[(k0 - 11) * 80 + j];
            Az[m].u[w] = pk(lo, hif);
        }
        // Layer2: A2[M=n][K] = d2_w[K-12][n] for K in [12,32), n<10 (zeros kill rows 0..11 of B)
        const float a2lo = (k0     >= 12 && v < 10) ? d2w[(k0 - 12) * 10 + v] : 0.0f;
        const float a2hi = (k0 + 1 >= 12 && v < 10) ? d2w[(k0 - 11) * 10 + v] : 0.0f;
        A2.u[w] = pk(a2lo, a2hi);
        // Layer3: A3[M=n][K=k] = k<10 ? (n<3 ? lc_w[k][n] : n==3 ? d1_w[k] : 0) : 0
        const float a3lo = (k0     < 10 && v < 4) ? ((v < 3) ? lcw[k0 * 3 + v]       : d1w[k0])     : 0.0f;
        const float a3hi = (k0 + 1 < 10 && v < 4) ? ((v < 3) ? lcw[(k0 + 1) * 3 + v] : d1w[k0 + 1]) : 0.0f;
        A3.u[w] = pk(a3lo, a3hi);
    }
#pragma unroll
    for (int m = 0; m < 5; ++m)
#pragma unroll
        for (int r = 0; r < 8; ++r) biasZ[m][r] = lb[16 * m + 8 * hi + r];
#pragma unroll
    for (int r = 0; r < 8; ++r) {
        const int u = r + 8 * hi;
        bias2[r] = (u < 10) ? d2b[u] : 0.0f;
        bias3[r] = (u < 3) ? lcb[u] : ((u == 3) ? d1b[0] : 0.0f);
    }

    // ---- State --------------------------------------------------------------
    float h[10], c[10];
#pragma unroll
    for (int j = 0; j < 10; ++j) {
        h[j] = h0[(size_t)vg * 20 + ub + j];
        c[j] = c0[(size_t)vg * 20 + ub + j];
    }
    float pos = init_state[(size_t)vg * 2 + 0];
    float spd = init_state[(size_t)vg * 2 + 1];

    // Pack h into Blds rows 12..31 (B layout: lane-half shifts K by 16, 2 K per word)
    auto pack_h = [&]() {
#pragma unroll
        for (int p = 0; p < 5; ++p) {
            const int k  = 12 + ub + 2 * p;             // even row pair (k,k+1)
            const int bl = v + ((k >= 16) ? 16 : 0);
            const int bw = (k & 15) >> 1;
            Blds[bl][bw] = pk(h[2 * p], h[2 * p + 1]);
        }
    };
    pack_h();

    const float* ipbase = inp + (size_t)vg * nt * 12 + 6 * hi;

    for (int t = 0; t < nt; ++t) {
        const float* ip = ipbase + (size_t)t * 12;
        __builtin_prefetch(ip + 12 * 16, 0, 1);   // global_prefetch_b8, 16 steps ahead

        // ---- Input preprocessing: half-wave 0 -> rows 0..5, half-wave 1 -> rows 6..11
        float r0, r1, r2, r3, r4, r5;
        {
            const float e0 = ip[0], e1 = ip[1], e2 = ip[2];
            const float e3 = ip[3], e4 = ip[4], e5 = ip[5];
            if (hi == 0) {
                r0 = (e0 - pos) * 0.01f; r1 = (e1 - pos) * 0.01f; r2 = (e2 - pos) * 0.01f;
                r3 = (pos - e3) * 0.01f; r4 = (pos - e4) * 0.01f; r5 = (pos - e5) * 0.01f;
            } else {
                r0 = e0 * 0.025f; r1 = e1 * 0.025f; r2 = e2 * 0.025f;
                r3 = e3 * 0.025f; r4 = e4 * 0.025f; r5 = e5 * 0.025f;
            }
        }
        r0 = nanfix(r0); r1 = nanfix(r1); r2 = nanfix(r2);
        r3 = nanfix(r3); r4 = nanfix(r4); r5 = nanfix(r5);
        {
            const int base = 3 * hi;  // hi0 -> words 0..2 (rows 0..5), hi1 -> words 3..5 (rows 6..11)
            Blds[v][base + 0] = pk(r0, r1);
            Blds[v][base + 1] = pk(r2, r3);
            Blds[v][base + 2] = pk(r4, r5);
        }

        // ---- LSTM GEMM: 5 x v_wmma_f32_16x16x32_f16, C seeded with bias ------
        V16H B;
#pragma unroll
        for (int w = 0; w < 8; ++w) B.u[w] = Blds[lane][w];
        v8f acc[5];
#pragma unroll
        for (int m = 0; m < 5; ++m)
            acc[m] = __builtin_amdgcn_wmma_f32_16x16x32_f16(
                false, Az[m].h, false, B.h, (short)0, biasZ[m], false, false);

        // Stage z^T to LDS: lane holds rows 16m+8hi..+7 of column v (contiguous in zlds[v])
#pragma unroll
        for (int m = 0; m < 5; ++m)
#pragma unroll
            for (int r = 0; r < 8; ++r)
                zlds[v][16 * m + 8 * hi + r] = acc[m][r];

        // ---- Gates (lane owns 10 units of its vehicle) -----------------------
#pragma unroll
        for (int j = 0; j < 10; ++j) {
            const int u = ub + j;
            const float zi = zlds[v][u],      zf = zlds[v][20 + u];
            const float zg = zlds[v][40 + u], zo = zlds[v][60 + u];
            const float cj = sigm(zf) * c[j] + sigm(zi) * tanh_f(zg);
            c[j] = cj;
            h[j] = sigm(zo) * tanh_f(cj);
        }
        pack_h();  // new h -> B rows 12..31 (also becomes next step's recurrent input)

        // ---- Layer 2: x^T = relu(d2_w^T @ h + b); reuses B (rows 0..11 zeroed by A2)
        V16H B2;
#pragma unroll
        for (int w = 0; w < 8; ++w) B2.u[w] = Blds[lane][w];
        v8f x8 = __builtin_amdgcn_wmma_f32_16x16x32_f16(
            false, A2.h, false, B2.h, (short)0, bias2, false, false);
#pragma unroll
        for (int r = 0; r < 8; ++r) {
            const int u = r + 8 * hi;
            xlds[v][u] = (u < 10) ? fmaxf(x8[r], 0.0f) : 0.0f;
        }

        // ---- Layer 3: [lc | acc_raw]^T = W3^T @ x ----------------------------
        V16H B3;
#pragma unroll
        for (int w = 0; w < 8; ++w) {
            const unsigned wv = pk(xlds[v][2 * w], xlds[v][2 * w + 1]); // K rows 0..15
            B3.u[w] = hi ? 0u : wv;                                     // K rows 16..31 = 0
        }
        v8f y = __builtin_amdgcn_wmma_f32_16x16x32_f16(
            false, A3.h, false, B3.h, (short)0, bias3, false, false);

        // ---- Integrate + emit (rows 0..3 live in lanes 0..15) ----------------
        const float acc_a = 10.0f * y[3] - 6.0f;  // (MAXA-MINA)*raw + MINA
        const float pn = pos + 0.1f * spd;
        const float sn = spd + 0.1f * acc_a;
        pos = pn; spd = sn;
        if (hi == 0) {
            out_pos[(size_t)vg * nt + t] = pn;
            float* lcp = out_lc + ((size_t)vg * nt + t) * 3;
            lcp[0] = y[0]; lcp[1] = y[1]; lcp[2] = y[2];
        }
    }

    // ---- Final state ---------------------------------------------------------
    if (hi == 0) out_spd[vg] = spd;
#pragma unroll
    for (int j = 0; j < 10; ++j) {
        out_h[(size_t)vg * 20 + ub + j] = h[j];
        out_c[(size_t)vg * 20 + ub + j] = c[j];
    }
}

extern "C" void kernel_launch(void* const* d_in, const int* in_sizes, int n_in,
                              void* d_out, int out_size, void* d_ws, size_t ws_size,
                              hipStream_t stream) {
    const float* inp        = (const float*)d_in[0];
    const float* init_state = (const float*)d_in[1];
    const float* h0         = (const float*)d_in[2];
    const float* c0         = (const float*)d_in[3];
    const float* lk         = (const float*)d_in[4];
    const float* lrk        = (const float*)d_in[5];
    const float* lb         = (const float*)d_in[6];
    const float* d2w        = (const float*)d_in[7];
    const float* d2b        = (const float*)d_in[8];
    const float* lcw        = (const float*)d_in[9];
    const float* lcb        = (const float*)d_in[10];
    const float* d1w        = (const float*)d_in[11];
    const float* d1b        = (const float*)d_in[12];

    const int nveh = in_sizes[1] / 2;
    const int nt   = in_sizes[0] / (nveh * 12);

    float* out     = (float*)d_out;
    float* out_pos = out;
    float* out_lc  = out_pos + (size_t)nveh * nt;
    float* out_spd = out_lc  + (size_t)nveh * nt * 3;
    float* out_h   = out_spd + nveh;
    float* out_c   = out_h   + (size_t)nveh * 20;

    dim3 grid(nveh / 16);   // one wave32 per 16-vehicle tile
    rnncf_kernel<<<grid, 32, 0, stream>>>(
        inp, init_state, h0, c0, lk, lrk, lb, d2w, d2b, lcw, lcb, d1w, d1b,
        out_pos, out_lc, out_spd, out_h, out_c, nveh, nt);
}
